// MGAT_24970939859571
// MI455X (gfx1250) — matrix operand order repaired and checked by reading, not compiled, not run on hardware
//
#include <hip/hip_runtime.h>

// ---------------- problem constants ----------------
#define Bq    4
#define Hg    128
#define Wgc   128
#define Nn    (Hg*Wgc)        // 16384
#define Kn    16
#define Dd    256
#define SDd   64
#define Mrows (Bq*Nn)         // 65536
#define BIG_NEG   -10000.0f
#define MGAT_R0SQ 2.25f       // 1.5^2
#define TV_W      0.001f
#define LN_EPS    1e-5f

#if __has_builtin(__builtin_amdgcn_global_load_async_to_lds_b128)
#define HAS_ASYNC_LDS 1
#else
#define HAS_ASYNC_LDS 0
#endif

// scales: (r, s0, sl, cb)
__device__ __constant__ float SC_R[3]  = {1.5f, 2.5f, 4.0f};
__device__ __constant__ float SC_S0[3] = {0.25f, 0.35f, 0.55f};
__device__ __constant__ float SC_SL[3] = {0.2f, 0.2f, 0.25f};
__device__ __constant__ float SC_CB[3] = {0.7f, 0.5f, 0.3f};

// ---------------- types ----------------
typedef __bf16 bf16_t;
typedef bf16_t v16bf __attribute__((ext_vector_type(16)));
typedef float  v8f   __attribute__((ext_vector_type(8)));
typedef int    i4v   __attribute__((ext_vector_type(4)));

union FragB { v16bf v; uint4 q[2]; };

__device__ __forceinline__ unsigned short f2b(float f) {
    unsigned int x = __float_as_uint(f);
    unsigned int r = x + 0x7fffu + ((x >> 16) & 1u);   // round-to-nearest-even
    return (unsigned short)(r >> 16);
}
__device__ __forceinline__ float b2f(unsigned short u) {
    return __uint_as_float(((unsigned int)u) << 16);
}
__device__ __forceinline__ float gelu_exact(float x) {
    return 0.5f * x * (1.0f + erff(x * 0.70710678118654752f));
}

// ---------------- 1) f32 -> bf16 convert ----------------
__global__ void cvt_bf16_kernel(const float* __restrict__ src,
                                unsigned short* __restrict__ dst, int n4) {
    int i = blockIdx.x * blockDim.x + threadIdx.x;
    if (i >= n4) return;
    float4 f = *(const float4*)(src + (size_t)i * 4);
    uint2 o;
    o.x = (unsigned int)f2b(f.x) | ((unsigned int)f2b(f.y) << 16);
    o.y = (unsigned int)f2b(f.z) | ((unsigned int)f2b(f.w) << 16);
    *(uint2*)(dst + (size_t)i * 4) = o;
}

// ---------------- 2) pack W (K=256 x Ncols) into WMMA-B fragment layout ---
// B frag (32x16 bf16): lane L holds col n = L&15; element e holds
// k = ((L>>4)<<3) + (e&7) + ((e>>3)<<4)  within a 32-deep K slab. One thread per (kt,nt,lane).
__global__ void pack_w_kernel(const float* __restrict__ W,
                              unsigned short* __restrict__ WP, int Ncols) {
    int ntN = Ncols >> 4;
    int tid = blockIdx.x * blockDim.x + threadIdx.x;
    int total = 8 * ntN * 32;
    if (tid >= total) return;
    int lane = tid & 31;
    int nt   = (tid >> 5) % ntN;
    int kt   = tid / (32 * ntN);
    int n    = (nt << 4) + (lane & 15);
    unsigned short* dst = WP + (size_t)tid * 16;
    #pragma unroll
    for (int e = 0; e < 16; ++e) {
        int k = kt * 32 + ((lane >> 4) << 3) + (e & 7) + ((e >> 3) << 4);
        dst[e] = f2b(W[(size_t)k * Ncols + n]);
    }
}

// ---------------- 3) WMMA GEMM: out[M x Ncols] = A[M x 256] * W + bias ----
// Wave tile = (TM*16) x (TN*16): TM A-fragments x TN B-fragments feed
// TM*TN independent WMMAs per 32-deep K step. A tile staged in LDS
// (async global->LDS on CDNA5); B pre-packed, streamed from global/L2.
template <int TM>
__device__ __forceinline__ void load_afrags(FragB* a, const unsigned short* As,
                                            int K0, int row, int half8) {
    #pragma unroll
    for (int mt = 0; mt < TM; ++mt) {
        const unsigned short* Abase = As + (mt * 16 + row) * 256;
        a[mt].q[0] = *(const uint4*)(Abase + K0 + half8);        // K = K0+half8..+7
        a[mt].q[1] = *(const uint4*)(Abase + K0 + 16 + half8);   // K = K0+16+half8..
    }
}

template <int TN>
__device__ __forceinline__ void load_bfrags(FragB* b, const unsigned short* __restrict__ WP,
                                            int kt, int ntN, int nt0, int lane) {
    const unsigned short* bp = WP + ((size_t)(kt * ntN + nt0) * 32 + lane) * 16;
    #pragma unroll
    for (int j = 0; j < TN; ++j) {
        b[j].q[0] = *(const uint4*)(bp + j * 512);
        b[j].q[1] = *(const uint4*)(bp + j * 512 + 8);
    }
}

template <int TM, int TN>
__device__ __forceinline__ void mma_step(v8f (*acc)[TN], const FragB* a, const FragB* b) {
    #pragma unroll
    for (int mt = 0; mt < TM; ++mt)
        #pragma unroll
        for (int j = 0; j < TN; ++j)
            acc[mt][j] = __builtin_amdgcn_wmma_f32_16x16x32_bf16(
                false, a[mt].v, false, b[j].v, (short)0, acc[mt][j], false, false);
}

template <int TM, int TN>
__global__ void __launch_bounds__(128)
gemm_wmma_kernel(const unsigned short* __restrict__ A,
                 const unsigned short* __restrict__ WP,
                 const float* __restrict__ bias,
                 const float* __restrict__ residual,
                 float* __restrict__ outf,
                 unsigned short* __restrict__ outb,
                 int Ncols, int act) {
    __shared__ unsigned short As[TM * 16 * 256];   // TM*8 KB A tile
    const int tid   = threadIdx.x;
    const int lane  = tid & 31;
    const int wave  = tid >> 5;
    const int nwav  = blockDim.x >> 5;
    const int m0    = blockIdx.x * (TM * 16);
    const int ntN   = Ncols >> 4;
    const int nt0   = (blockIdx.y * nwav + wave) * TN;

    // ---- stage A tile (TM*512 x b128), async to LDS when available
#if HAS_ASYNC_LDS
    for (int t = tid; t < TM * 512; t += blockDim.x) {
        int row = (t * 8) >> 8;
        int col = (t * 8) & 255;
        __builtin_amdgcn_global_load_async_to_lds_b128(
            (__attribute__((address_space(1))) i4v*)(A + (size_t)(m0 + row) * 256 + col),
            (__attribute__((address_space(3))) i4v*)(As + t * 8),
            0, 0);
    }
#if __has_builtin(__builtin_amdgcn_s_wait_asynccnt)
    __builtin_amdgcn_s_wait_asynccnt(0);
#else
    asm volatile("s_wait_asynccnt 0x0" ::: "memory");
#endif
#else
    for (int t = tid; t < TM * 512; t += blockDim.x) {
        int row = (t * 8) >> 8;
        int col = (t * 8) & 255;
        ((uint4*)As)[t] = *(const uint4*)(A + (size_t)(m0 + row) * 256 + col);
    }
#endif
    __syncthreads();

    v8f acc[TM][TN];
    #pragma unroll
    for (int mt = 0; mt < TM; ++mt)
        #pragma unroll
        for (int j = 0; j < TN; ++j) acc[mt][j] = (v8f){};

    const int row   = lane & 15;
    const int half8 = (lane >> 4) << 3;

    // ---- double-buffered K loop (8 steps of K=32)
    FragB a0[TM], a1[TM], b0[TN], b1[TN];
    load_afrags<TM>(a0, As, 0, row, half8);
    load_bfrags<TN>(b0, WP, 0, ntN, nt0, lane);
    #pragma unroll
    for (int kt = 0; kt < 8; kt += 2) {
        load_afrags<TM>(a1, As, (kt + 1) * 32, row, half8);
        load_bfrags<TN>(b1, WP, kt + 1, ntN, nt0, lane);
        mma_step<TM, TN>(acc, a0, b0);
        if (kt + 2 < 8) {
            load_afrags<TM>(a0, As, (kt + 2) * 32, row, half8);
            load_bfrags<TN>(b0, WP, kt + 2, ntN, nt0, lane);
        }
        mma_step<TM, TN>(acc, a1, b1);
    }

    // ---- epilogue: lane L holds col (L&15); VGPR r -> row r + ((L>>4)<<3)
    #pragma unroll
    for (int mt = 0; mt < TM; ++mt) {
        const int rbase = m0 + mt * 16 + ((lane >> 4) << 3);
        #pragma unroll
        for (int j = 0; j < TN; ++j) {
            const int col = ((nt0 + j) << 4) + (lane & 15);
            const float bv = bias ? bias[col] : 0.0f;
            #pragma unroll
            for (int r = 0; r < 8; ++r) {
                const int grow = rbase + r;
                float v = acc[mt][j][r] + bv;
                if (residual) v += residual[(size_t)grow * Ncols + col];
                if (act) v = gelu_exact(v);
                if (outf) outf[(size_t)grow * Ncols + col] = v;
                else      outb[(size_t)grow * Ncols + col] = f2b(v);
            }
        }
    }
}

// ---------------- 4) fused neighbor attention (one wave per node) --------
__global__ void attn_kernel(const unsigned short* __restrict__ qb,
                            const unsigned short* __restrict__ kb,
                            const unsigned short* __restrict__ vb,
                            const float* __restrict__ S,
                            const float* __restrict__ zembp,
                            const float* __restrict__ rel,
                            const float* __restrict__ geod,
                            const int*   __restrict__ idx,
                            const float* __restrict__ Ws1, const float* __restrict__ bs1,
                            const float* __restrict__ Ws2, const float* __restrict__ bs2,
                            const float* __restrict__ Wph, const float* __restrict__ bph,
                            const float* __restrict__ Wsg, const float* __restrict__ bsg,
                            unsigned short* __restrict__ aggb,
                            float* __restrict__ anorm) {
    const int lane = threadIdx.x & 31;
    const int wave = threadIdx.x >> 5;
    const int node = blockIdx.x * (blockDim.x >> 5) + wave;   // 0..Mrows-1
    const int b  = node >> 14;          // /N (16384)
    const int nl = node & (Nn - 1);

    // q row: 8 channels per lane, bf16 -> f32
    float qf[8];
    {
        uint4 qv = *(const uint4*)(qb + (size_t)node * Dd + lane * 8);
        const unsigned short* qs = (const unsigned short*)&qv;
        #pragma unroll
        for (int c = 0; c < 8; ++c) qf[c] = b2f(qs[c]);
    }

    // S stats (m1, m2) across SD=64: 2 elems per lane
    float2 sOwn = *(const float2*)(S + (size_t)node * SDd + lane * 2);
    float sum = sOwn.x + sOwn.y;
    float sq  = sOwn.x * sOwn.x + sOwn.y * sOwn.y;
    #pragma unroll
    for (int o = 16; o; o >>= 1) { sum += __shfl_xor(sum, o); sq += __shfl_xor(sq, o); }
    const float m1 = sum * (1.0f / SDd);
    const float m2 = sq  * (1.0f / SDd);

    // s_emb layer 1 (all lanes, cheap): 2 -> 16
    float h1[16];
    #pragma unroll
    for (int j = 0; j < 16; ++j)
        h1[j] = gelu_exact(m1 * Ws1[j] + m2 * Ws1[16 + j] + bs1[j]);

    // layer 2 lane-distributed: lane u<16 owns unit u
    const int u = lane & 15;
    float h2u = bs2[u];
    #pragma unroll
    for (int j = 0; j < 16; ++j) h2u += h1[j] * Ws2[j * 16 + u];
    h2u = gelu_exact(h2u);
    const float zeu  = zembp[(size_t)node * 16 + u];    // gelu already applied in GEMM
    const float msk  = (lane < 16) ? 1.0f : 0.0f;

    // abc = ker_in @ Wph + bph ; glog = ker_in @ Wsg + bsg (3 outputs each)
    float abc[3], gl[3];
    #pragma unroll
    for (int t = 0; t < 3; ++t) {
        float pa = msk * (h2u * Wph[u * 3 + t] + zeu * Wph[(16 + u) * 3 + t]);
        float pg = msk * (h2u * Wsg[u * 3 + t] + zeu * Wsg[(16 + u) * 3 + t]);
        #pragma unroll
        for (int o = 16; o; o >>= 1) { pa += __shfl_xor(pa, o); pg += __shfl_xor(pg, o); }
        abc[t] = pa + bph[t];
        gl[t]  = pg + bsg[t];
    }
    // gate softmax
    float gm = fmaxf(gl[0], fmaxf(gl[1], gl[2]));
    float g0 = expf(gl[0] - gm), g1 = expf(gl[1] - gm), g2 = expf(gl[2] - gm);
    float gs = 1.0f / (g0 + g1 + g2);
    const float gw[3] = {g0 * gs, g1 * gs, g2 * gs};

    const float La = expf(abc[0]), Lb = abc[1], Lc = expf(abc[2]);
    const float A00 = La * La, A01 = La * Lb, A11 = Lb * Lb + Lc * Lc;
    if (lane == 0)
        anorm[node] = sqrtf(A00 * A00 + 2.0f * A01 * A01 + A11 * A11);

    // online softmax over K=16 neighbors
    float mrun = -INFINITY, ssum = 0.0f;
    float aggc[8] = {0, 0, 0, 0, 0, 0, 0, 0};

    for (int k = 0; k < Kn; ++k) {
        const int nb = idx[nl * Kn + k];
        const size_t gi = (size_t)b * Nn + nb;

        // content + sim partials
        uint4 kv = *(const uint4*)(kb + gi * Dd + lane * 8);
        const unsigned short* ks = (const unsigned short*)&kv;
        float cp = 0.0f;
        #pragma unroll
        for (int c = 0; c < 8; ++c) cp += qf[c] * b2f(ks[c]);
        float2 sn = *(const float2*)(S + gi * SDd + lane * 2);
        float sp = sOwn.x * sn.x + sOwn.y * sn.y;
        #pragma unroll
        for (int o = 16; o; o >>= 1) { cp += __shfl_xor(cp, o); sp += __shfl_xor(sp, o); }
        const float content = cp * 0.0625f;   // 1/sqrt(256)

        const float r0 = rel[(nl * Kn + k) * 2 + 0];
        const float r1 = rel[(nl * Kn + k) * 2 + 1];
        const float gd = geod[nl * Kn + k];
        const float rAr = A00 * r0 * r0 + 2.0f * A01 * r0 * r1 + A11 * r1 * r1;
        const float cprior = expf(-gd * gd / MGAT_R0SQ);

        float logit = content + sp - 0.25f * gd;   // terms common to all scales (sum g = 1)
        #pragma unroll
        for (int i = 0; i < 3; ++i) {
            const float sg  = SC_S0[i] + SC_SL[i] * gd;
            const float s2v = fmaxf(sg * sg, 1e-6f);
            float li = -rAr / s2v + SC_CB[i] * cprior;
            if (gd > SC_R[i]) li += BIG_NEG;
            logit += gw[i] * li;
        }

        const float mnew  = fmaxf(mrun, logit);
        const float scale = expf(mrun - mnew);
        const float w     = expf(logit - mnew);
        ssum = ssum * scale + w;

        uint4 vv = *(const uint4*)(vb + gi * Dd + lane * 8);
        const unsigned short* vs = (const unsigned short*)&vv;
        #pragma unroll
        for (int c = 0; c < 8; ++c) aggc[c] = aggc[c] * scale + w * b2f(vs[c]);
        mrun = mnew;
    }

    const float inv = 1.0f / ssum;
    unsigned short ob[8];
    #pragma unroll
    for (int c = 0; c < 8; ++c) ob[c] = f2b(aggc[c] * inv);
    uint4 ov;
    unsigned int* op = (unsigned int*)&ov;
    op[0] = (unsigned int)ob[0] | ((unsigned int)ob[1] << 16);
    op[1] = (unsigned int)ob[2] | ((unsigned int)ob[3] << 16);
    op[2] = (unsigned int)ob[4] | ((unsigned int)ob[5] << 16);
    op[3] = (unsigned int)ob[6] | ((unsigned int)ob[7] << 16);
    *(uint4*)(aggb + (size_t)node * Dd + lane * 8) = ov;
}

// ---------------- 5) in-place LayerNorm over D=256 (wave per row) --------
__global__ void ln_kernel(float* __restrict__ h,
                          const float* __restrict__ g,
                          const float* __restrict__ bta) {
    const int lane = threadIdx.x & 31;
    const int wave = threadIdx.x >> 5;
    const int row  = blockIdx.x * (blockDim.x >> 5) + wave;
    float* p = h + (size_t)row * Dd + lane * 8;
    float4 a = *(const float4*)p;
    float4 c = *(const float4*)(p + 4);
    float s  = a.x + a.y + a.z + a.w + c.x + c.y + c.z + c.w;
    float sq = a.x * a.x + a.y * a.y + a.z * a.z + a.w * a.w +
               c.x * c.x + c.y * c.y + c.z * c.z + c.w * c.w;
    #pragma unroll
    for (int o = 16; o; o >>= 1) { s += __shfl_xor(s, o); sq += __shfl_xor(sq, o); }
    const float mu  = s * (1.0f / Dd);
    const float var = sq * (1.0f / Dd) - mu * mu;
    const float inv = rsqrtf(var + LN_EPS);
    const int cb = lane * 8;
    float x[8] = {a.x, a.y, a.z, a.w, c.x, c.y, c.z, c.w};
    #pragma unroll
    for (int j = 0; j < 8; ++j) x[j] = g[cb + j] * (x[j] - mu) * inv + bta[cb + j];
    *(float4*)p       = make_float4(x[0], x[1], x[2], x[3]);
    *(float4*)(p + 4) = make_float4(x[4], x[5], x[6], x[7]);
}

// ---------------- 6) TV regularizer, two-stage deterministic reduction ---
__global__ void tv_partial_kernel(const float* __restrict__ an,
                                  float* __restrict__ part) {
    __shared__ float sdx[256], sdy[256];
    const int tid = threadIdx.x;
    const int i = blockIdx.x * 256 + tid;   // over B*H*W = 65536
    float dx = 0.0f, dy = 0.0f;
    const int c = i & (Wgc - 1), r = (i >> 7) & (Hg - 1);
    const float a0 = an[i];
    if (c < Wgc - 1) dx = fabsf(an[i + 1] - a0);
    if (r < Hg - 1)  dy = fabsf(an[i + Wgc] - a0);
    sdx[tid] = dx; sdy[tid] = dy;
    __syncthreads();
    for (int o = 128; o; o >>= 1) {
        if (tid < o) { sdx[tid] += sdx[tid + o]; sdy[tid] += sdy[tid + o]; }
        __syncthreads();
    }
    if (tid == 0) { part[blockIdx.x * 2] = sdx[0]; part[blockIdx.x * 2 + 1] = sdy[0]; }
}

__global__ void tv_final_kernel(const float* __restrict__ part,
                                float* __restrict__ out) {
    __shared__ float sdx[256], sdy[256];
    const int t = threadIdx.x;
    sdx[t] = part[t * 2]; sdy[t] = part[t * 2 + 1];
    __syncthreads();
    for (int o = 128; o; o >>= 1) {
        if (t < o) { sdx[t] += sdx[t + o]; sdy[t] += sdy[t + o]; }
        __syncthreads();
    }
    if (t == 0) {
        const float dxm = sdx[0] / (float)(Bq * Hg * (Wgc - 1));
        const float dym = sdy[0] / (float)(Bq * (Hg - 1) * Wgc);
        out[0] = TV_W * (dxm + dym);
    }
}

// ---------------- host launch ----------------
extern "C" void kernel_launch(void* const* d_in, const int* in_sizes, int n_in,
                              void* d_out, int out_size, void* d_ws, size_t ws_size,
                              hipStream_t stream) {
    const float* z    = (const float*)d_in[0];
    const float* S    = (const float*)d_in[1];
    const float* rel  = (const float*)d_in[2];
    const float* geod = (const float*)d_in[3];
    const int*   idx  = (const int*)d_in[4];
    const float* Wq = (const float*)d_in[5],  *bq = (const float*)d_in[6];
    const float* Wk = (const float*)d_in[7],  *bk = (const float*)d_in[8];
    const float* Wv = (const float*)d_in[9],  *bv = (const float*)d_in[10];
    const float* Wo = (const float*)d_in[11], *bo = (const float*)d_in[12];
    const float* ln_g = (const float*)d_in[13], *ln_b = (const float*)d_in[14];
    const float* Ws1 = (const float*)d_in[15], *bs1 = (const float*)d_in[16];
    const float* Ws2 = (const float*)d_in[17], *bs2 = (const float*)d_in[18];
    const float* Wzg = (const float*)d_in[19], *bzg = (const float*)d_in[20];
    const float* Wph = (const float*)d_in[21], *bph = (const float*)d_in[22];
    const float* Wsg = (const float*)d_in[23], *bsg = (const float*)d_in[24];

    // workspace carve-up
    char* base = (char*)d_ws;
    size_t off = 0;
    auto carve = [&](size_t bytes) {
        void* p = base + off;
        off += (bytes + 255) & ~(size_t)255;
        return p;
    };
    const size_t MD = (size_t)Mrows * Dd;
    unsigned short* zb    = (unsigned short*)carve(MD * 2);   // aliased as aggb later
    unsigned short* qb    = (unsigned short*)carve(MD * 2);
    unsigned short* kb    = (unsigned short*)carve(MD * 2);
    unsigned short* vb    = (unsigned short*)carve(MD * 2);
    float*          zembp = (float*)carve((size_t)Mrows * 16 * 4);
    unsigned short* WqP   = (unsigned short*)carve((size_t)Dd * Dd * 2);
    unsigned short* WkP   = (unsigned short*)carve((size_t)Dd * Dd * 2);
    unsigned short* WvP   = (unsigned short*)carve((size_t)Dd * Dd * 2);
    unsigned short* WoP   = (unsigned short*)carve((size_t)Dd * Dd * 2);
    unsigned short* WzgP  = (unsigned short*)carve((size_t)Dd * 16 * 2);
    float*          anorm = (float*)carve((size_t)Mrows * 4);
    float*          parts = (float*)carve(256 * 2 * 4);
    unsigned short* aggb  = zb;   // z bf16 is dead once q/k/v/z_emb GEMMs finish

    float* outF = (float*)d_out;
    float* regF = outF + MD;

    // 1) convert z to bf16
    {
        int n4 = (int)(MD / 4);
        cvt_bf16_kernel<<<(n4 + 255) / 256, 256, 0, stream>>>(z, zb, n4);
    }
    // 2) pack weights into WMMA-B layout
    pack_w_kernel<<<16, 256, 0, stream>>>(Wq,  WqP,  Dd);
    pack_w_kernel<<<16, 256, 0, stream>>>(Wk,  WkP,  Dd);
    pack_w_kernel<<<16, 256, 0, stream>>>(Wv,  WvP,  Dd);
    pack_w_kernel<<<16, 256, 0, stream>>>(Wo,  WoP,  Dd);
    pack_w_kernel<<<1,  256, 0, stream>>>(Wzg, WzgP, 16);

    // 3) WMMA GEMMs: q, k, v (bf16 out), z_emb (f32 + gelu)
    dim3 gBig(Mrows / 32, 1);     // block = 32 rows x 256 cols (4 waves, 32x64/wave)
    gemm_wmma_kernel<2, 4><<<gBig, 128, 0, stream>>>(zb, WqP, bq, nullptr, nullptr, qb, Dd, 0);
    gemm_wmma_kernel<2, 4><<<gBig, 128, 0, stream>>>(zb, WkP, bk, nullptr, nullptr, kb, Dd, 0);
    gemm_wmma_kernel<2, 4><<<gBig, 128, 0, stream>>>(zb, WvP, bv, nullptr, nullptr, vb, Dd, 0);
    dim3 gSmall(Mrows / 16, 1);
    gemm_wmma_kernel<1, 1><<<gSmall, 32, 0, stream>>>(zb, WzgP, bzg, nullptr, zembp, nullptr, 16, 1);

    // 4) fused attention (8 waves/block, one wave per node)
    attn_kernel<<<Mrows / 8, 256, 0, stream>>>(qb, kb, vb, S, zembp, rel, geod, idx,
                                               Ws1, bs1, Ws2, bs2, Wph, bph, Wsg, bsg,
                                               aggb, anorm);

    // 5) output projection with residual: h = z + agg @ Wo + bo  -> d_out (f32)
    gemm_wmma_kernel<2, 4><<<gBig, 128, 0, stream>>>(aggb, WoP, bo, z, outF, nullptr, Dd, 0);

    // 6) in-place LayerNorm
    ln_kernel<<<Mrows / 8, 256, 0, stream>>>(outF, ln_g, ln_b);

    // 7) TV regularizer scalar
    tv_partial_kernel<<<256, 256, 0, stream>>>(anorm, parts);
    tv_final_kernel<<<1, 256, 0, stream>>>(parts, regF);
}
